// LatexDecoderHeader_25434796327413
// MI455X (gfx1250) — matrix operand order repaired and checked by reading, not compiled, not run on hardware
//
#include <hip/hip_runtime.h>
#include <hip/hip_bf16.h>
#include <math.h>

typedef __bf16 bf16;
typedef __attribute__((ext_vector_type(8)))  bf16  bf16x8;
typedef __attribute__((ext_vector_type(16))) bf16  bf16x16;
typedef __attribute__((ext_vector_type(8)))  float f32x8;

#define WMMA_BF16(a, b, c) \
  __builtin_amdgcn_wmma_f32_16x16x32_bf16(false, (a), false, (b), (short)0, (c), false, false)

static __device__ __forceinline__ bf16x16 join16(bf16x8 lo, bf16x8 hi) {
  return __builtin_shufflevector(lo, hi, 0,1,2,3,4,5,6,7,8,9,10,11,12,13,14,15);
}

// ---------------------------------------------------------------------------
// Generic bf16 WMMA GEMM: C[m,n] = sum_k A[m,k] * B[n,k]   (B rows are weight rows)
// One wave per 16x16 output tile. A,B row-major bf16; C f32.
// act: 0 = none, 1 = tanh
// ---------------------------------------------------------------------------
__global__ __launch_bounds__(256) void gemm_bf16(
    const bf16* __restrict__ A, int lda,
    const bf16* __restrict__ B, int ldb,
    float* __restrict__ C, int ldc,
    int Mtiles, int Ntiles, int K, int nTrue, int act)
{
  int wave = threadIdx.x >> 5;
  int lane = threadIdx.x & 31;
  int tile = blockIdx.x * 8 + wave;
  if (tile >= Mtiles * Ntiles) return;
  int mt = tile / Ntiles, nt = tile % Ntiles;
  int half = lane >> 4, lq = lane & 15;

  const bf16* rowA = A + (size_t)(mt * 16 + lq) * lda;
  int nrow = nt * 16 + lq; if (nrow >= nTrue) nrow = nTrue - 1;
  const bf16* rowB = B + (size_t)nrow * ldb;

  f32x8 acc = {};
  for (int k = 0; k < K; k += 32) {
    // A fragment: lanes 0-15 -> K = k..k+7 and k+16..k+23; lanes 16-31 -> +8
    int ka = k + half * 8;
    bf16x8 a_lo = *(const bf16x8*)(rowA + ka);
    bf16x8 a_hi = *(const bf16x8*)(rowA + ka + 16);
    // B fragment: same K layout as A
    int kb = k + half * 8;
    bf16x8 b_lo = *(const bf16x8*)(rowB + kb);
    bf16x8 b_hi = *(const bf16x8*)(rowB + kb + 16);
    acc = WMMA_BF16(join16(a_lo, a_hi), join16(b_lo, b_hi), acc);
  }
  int n = nt * 16 + lq;
  if (n < nTrue) {
#pragma unroll
    for (int e = 0; e < 8; ++e) {
      int m = mt * 16 + e + half * 8;
      float v = acc[e];
      if (act) v = tanhf(v);
      C[(size_t)m * ldc + n] = v;
    }
  }
}

// ---------------------------------------------------------------------------
// Fused attention-scores kernel (the 34 GFLOP GEMM):
// scores[b,l] = sum_d beta[d] * tanh( sum_k enc[b,l,k]*W1[d,k] + hW2[b,d] )
// W1 converted to bf16 in padded LDS (256 x 264). enc streamed f32 -> bf16 regs.
// One wave per 16 (b,l)-rows; block = 8 waves.
// ---------------------------------------------------------------------------
__global__ __launch_bounds__(256) void attn_scores(
    const float* __restrict__ enc, const float* __restrict__ W1,
    const float* __restrict__ hW2, const float* __restrict__ beta,
    float* __restrict__ scores)
{
  extern __shared__ bf16 w1s[];            // 256 rows x 264 (pad kills bank conflicts)
  const int STRIDE = 264;
  int tid = threadIdx.x;
  for (int e = tid; e < 256 * 256; e += 256) {
    int n = e >> 8, k = e & 255;
    w1s[n * STRIDE + k] = (bf16)W1[e];
  }
  __syncthreads();

  int wave = tid >> 5, lane = tid & 31;
  int half = lane >> 4, lq = lane & 15;
  int T  = blockIdx.x * 8 + wave;          // 16384 row-tiles total
  int b  = T >> 6;
  int lt = T & 63;

  const float* rowA = enc + ((size_t)b * 1024 + lt * 16 + lq) * 256;

  // Preload all 8 A fragments (K = 256) once; reuse across 16 N-tiles.
  bf16x16 afr[8];
#pragma unroll
  for (int kt = 0; kt < 8; ++kt) {
    int k0 = kt * 32 + half * 8;
    float4 f0 = *(const float4*)(rowA + k0);
    float4 f1 = *(const float4*)(rowA + k0 + 4);
    float4 f2 = *(const float4*)(rowA + k0 + 16);
    float4 f3 = *(const float4*)(rowA + k0 + 20);
    bf16x16 a;
    a[0]=(bf16)f0.x; a[1]=(bf16)f0.y; a[2]=(bf16)f0.z;  a[3]=(bf16)f0.w;
    a[4]=(bf16)f1.x; a[5]=(bf16)f1.y; a[6]=(bf16)f1.z;  a[7]=(bf16)f1.w;
    a[8]=(bf16)f2.x; a[9]=(bf16)f2.y; a[10]=(bf16)f2.z; a[11]=(bf16)f2.w;
    a[12]=(bf16)f3.x;a[13]=(bf16)f3.y;a[14]=(bf16)f3.z; a[15]=(bf16)f3.w;
    afr[kt] = a;
  }

  float partial[8] = {0,0,0,0,0,0,0,0};
  for (int nt = 0; nt < 16; ++nt) {
    float hv = hW2[b * 256 + nt * 16 + lq];
    float bv = beta[nt * 16 + lq];
    const bf16* brow = w1s + (nt * 16 + lq) * STRIDE;
    f32x8 acc = {};
#pragma unroll
    for (int kt = 0; kt < 8; ++kt) {
      int kb = kt * 32 + half * 8;
      bf16x8 b_lo = *(const bf16x8*)(brow + kb);
      bf16x8 b_hi = *(const bf16x8*)(brow + kb + 16);
      acc = WMMA_BF16(afr[kt], join16(b_lo, b_hi), acc);
    }
#pragma unroll
    for (int e = 0; e < 8; ++e)
      partial[e] += bv * tanhf(acc[e] + hv);
  }
  // Reduce over the 16 lanes of each half (each half holds rows m=e(+8), cols n)
#pragma unroll
  for (int msk = 8; msk >= 1; msk >>= 1)
#pragma unroll
    for (int e = 0; e < 8; ++e)
      partial[e] += __shfl_xor(partial[e], msk, 32);
  if (lq == 0) {
#pragma unroll
    for (int e = 0; e < 8; ++e)
      scores[(size_t)b * 1024 + lt * 16 + e + half * 8] = partial[e];
  }
}

// ---------------------------------------------------------------------------
__global__ __launch_bounds__(256) void softmax_rows(float* __restrict__ x, int cols)
{
  __shared__ float red[256];
  float* p = x + (size_t)blockIdx.x * cols;
  int t = threadIdx.x;
  float m = -INFINITY;
  for (int c = t; c < cols; c += 256) m = fmaxf(m, p[c]);
  red[t] = m; __syncthreads();
  for (int s = 128; s > 0; s >>= 1) { if (t < s) red[t] = fmaxf(red[t], red[t + s]); __syncthreads(); }
  m = red[0]; __syncthreads();
  float sum = 0.0f;
  for (int c = t; c < cols; c += 256) { float e = expf(p[c] - m); p[c] = e; sum += e; }
  red[t] = sum; __syncthreads();
  for (int s = 128; s > 0; s >>= 1) { if (t < s) red[t] += red[t + s]; __syncthreads(); }
  float inv = 1.0f / red[0];
  for (int c = t; c < cols; c += 256) p[c] *= inv;
}

// ---------------------------------------------------------------------------
__global__ __launch_bounds__(256) void lstm_cell(
    const float* __restrict__ gates, const float* __restrict__ bias,
    const float* __restrict__ c0, float* __restrict__ ht_out,
    float* __restrict__ ct_out, bf16* __restrict__ ycat)
{
  int idx = blockIdx.x * 256 + threadIdx.x;      // 65536
  int b = idx >> 8, h = idx & 255;
  const float* g = gates + (size_t)b * 1024;
  float ig = g[h]       + bias[h];
  float fg = g[256 + h] + bias[256 + h];
  float gg = g[512 + h] + bias[512 + h];
  float og = g[768 + h] + bias[768 + h];
  float si = 1.0f / (1.0f + expf(-ig));
  float sf = 1.0f / (1.0f + expf(-fg));
  float so = 1.0f / (1.0f + expf(-og));
  float c  = sf * c0[idx] + si * tanhf(gg);
  float ht = so * tanhf(c);
  ct_out[idx] = c;
  ht_out[idx] = ht;
  ycat[(size_t)b * 512 + h] = (bf16)ht;          // Ycat[:, 0:256] = h_t (bf16)
}

// context[b,d] = sum_l attn[b,l] * enc[b,l,d]  -> Ycat[:, 256:512] (bf16)
__global__ __launch_bounds__(256) void context_kernel(
    const float* __restrict__ attn, const float* __restrict__ enc,
    bf16* __restrict__ ycat)
{
  __shared__ float sa[1024];
  int b = blockIdx.x, t = threadIdx.x;
  for (int l = t; l < 1024; l += 256) sa[l] = attn[(size_t)b * 1024 + l];
  __syncthreads();
  const float* eb = enc + (size_t)b * 1024 * 256 + t;
  float acc = 0.0f;
#pragma unroll 4
  for (int l = 0; l < 1024; ++l) acc += sa[l] * eb[(size_t)l * 256];
  ycat[(size_t)b * 512 + 256 + t] = (bf16)acc;
}

// ---------------------------------------------------------------------------
__global__ __launch_bounds__(256) void prep_xcat(
    const int* __restrict__ lt, const float* __restrict__ emb,
    const float* __restrict__ o_t, const float* __restrict__ h0,
    bf16* __restrict__ x)
{
  int idx = blockIdx.x * 256 + threadIdx.x;      // 256*640
  if (idx >= 256 * 640) return;
  int b = idx / 640, c = idx % 640;
  float v;
  if      (c < 128) v = emb[(size_t)lt[b] * 128 + c];
  else if (c < 384) v = o_t[(size_t)b * 256 + (c - 128)];
  else              v = h0 [(size_t)b * 256 + (c - 384)];
  x[idx] = (bf16)v;
}

__global__ __launch_bounds__(256) void prep_wcat(
    const float* __restrict__ wih, const float* __restrict__ whh, bf16* __restrict__ w)
{
  int idx = blockIdx.x * 256 + threadIdx.x;      // 1024*640
  if (idx >= 1024 * 640) return;
  int j = idx / 640, c = idx % 640;
  w[idx] = (bf16)(c < 384 ? wih[(size_t)j * 384 + c] : whh[(size_t)j * 256 + (c - 384)]);
}

__global__ __launch_bounds__(256) void add_bias(
    const float* __restrict__ a, const float* __restrict__ b, float* __restrict__ o)
{
  int i = blockIdx.x * 256 + threadIdx.x;
  if (i < 1024) o[i] = a[i] + b[i];
}

__global__ __launch_bounds__(256) void cvt_bf16(
    const float* __restrict__ s, bf16* __restrict__ d, int n)
{
  int i = blockIdx.x * 256 + threadIdx.x;
  if (i < n) d[i] = (bf16)s[i];
}

// ---------------------------------------------------------------------------
extern "C" void kernel_launch(void* const* d_in, const int* in_sizes, int n_in,
                              void* d_out, int out_size, void* d_ws, size_t ws_size,
                              hipStream_t stream) {
  const float* h0    = (const float*)d_in[0];
  const float* c0    = (const float*)d_in[1];
  const float* o_t   = (const float*)d_in[2];
  const float* enc   = (const float*)d_in[3];
  const int*   lt    = (const int*)  d_in[4];
  const float* emb   = (const float*)d_in[5];
  const float* W_ih  = (const float*)d_in[6];
  const float* W_hh  = (const float*)d_in[7];
  const float* b_ih  = (const float*)d_in[8];
  const float* b_hh  = (const float*)d_in[9];
  const float* W1    = (const float*)d_in[10];
  const float* W2    = (const float*)d_in[11];
  const float* W3    = (const float*)d_in[12];
  const float* W_out = (const float*)d_in[13];
  const float* beta  = (const float*)d_in[14];

  float* out    = (float*)d_out;
  float* ht_o   = out;                 // 256*256
  float* ct_o   = out + 65536;         // 256*256
  float* onew_o = out + 131072;        // 256*256
  float* logit  = out + 196608;        // 256*172

  char* w = (char*)d_ws;
  bf16*  Xcat   = (bf16*) (w + 0);        // 256x640   bf16
  bf16*  Wcat   = (bf16*) (w + 327680);   // 1024x640  bf16
  float* gates  = (float*)(w + 1638400);  // 256x1024  f32
  bf16*  Ycat   = (bf16*) (w + 2686976);  // 256x512   bf16 (h_t | context)
  bf16*  W2b    = (bf16*) (w + 2949120);  // 256x256   bf16
  float* hW2    = (float*)(w + 3080192);  // 256x256   f32
  float* scores = (float*)(w + 3342336);  // 256x1024  f32
  bf16*  W3b    = (bf16*) (w + 4390912);  // 256x512   bf16
  bf16*  Woutb  = (bf16*) (w + 4653056);  // 172x256   bf16
  bf16*  onb    = (bf16*) (w + 4741120);  // 256x256   bf16
  float* bias   = (float*)(w + 4872192);  // 1024      f32

  // prep / conversions
  prep_xcat<<<640, 256, 0, stream>>>(lt, emb, o_t, h0, Xcat);
  prep_wcat<<<2560, 256, 0, stream>>>(W_ih, W_hh, Wcat);
  add_bias<<<4, 256, 0, stream>>>(b_ih, b_hh, bias);
  cvt_bf16<<<256, 256, 0, stream>>>(W2, W2b, 65536);
  cvt_bf16<<<512, 256, 0, stream>>>(W3, W3b, 131072);
  cvt_bf16<<<172, 256, 0, stream>>>(W_out, Woutb, 44032);

  // gates = Xcat @ Wcat^T  (M=256, N=1024, K=640)
  gemm_bf16<<<128, 256, 0, stream>>>(Xcat, 640, Wcat, 640, gates, 1024, 16, 64, 640, 1024, 0);
  // LSTM cell -> h_t, c_t, Ycat[:,0:256]
  lstm_cell<<<256, 256, 0, stream>>>(gates, bias, c0, ht_o, ct_o, Ycat);
  // hW2 = h_t @ W2^T  (M=256, N=256, K=256)
  gemm_bf16<<<32, 256, 0, stream>>>(Ycat, 512, W2b, 256, hW2, 256, 16, 16, 256, 256, 0);
  // scores (dominant fused GEMM), LDS = 256*264*2 bytes
  attn_scores<<<2048, 256, 256 * 264 * 2, stream>>>(enc, W1, hW2, beta, scores);
  // attn = softmax(scores) in place
  softmax_rows<<<256, 256, 0, stream>>>(scores, 1024);
  // context -> Ycat[:,256:512]
  context_kernel<<<256, 256, 0, stream>>>(scores, enc, Ycat);
  // o_new = tanh(Ycat @ W3^T)  (M=256, N=256, K=512)
  gemm_bf16<<<32, 256, 0, stream>>>(Ycat, 512, W3b, 512, onew_o, 256, 16, 16, 512, 256, 1);
  cvt_bf16<<<256, 256, 0, stream>>>(onew_o, onb, 65536);
  // logits = o_new @ W_out^T  (M=256, N=172 padded to 176, K=256)
  gemm_bf16<<<22, 256, 0, stream>>>(onb, 256, Woutb, 256, logit, 172, 16, 11, 256, 172, 0);
  softmax_rows<<<256, 256, 0, stream>>>(logit, 172);
}
